// VolumeSDFRenderer_43817256354158
// MI455X (gfx1250) — matrix operand order repaired and checked by reading, not compiled, not run on hardware
//
#include <hip/hip_runtime.h>
#include <hip/hip_bf16.h>

typedef __attribute__((ext_vector_type(16))) _Float16 v16h;
typedef __attribute__((ext_vector_type(8)))  float    v8f;

#define SDF_S       20.0f
#define NSAMP       128
#define LDS_P       131     // padded LDS row stride in floats (odd, mod 8 == 3)
#define RAYS_PER_WAVE   16
#define WAVES_PER_BLOCK 8
#define RAYS_PER_BLOCK  (RAYS_PER_WAVE * WAVES_PER_BLOCK)

static __device__ __forceinline__ v8f wmma16x16x32(v16h a, v16h b, v8f c) {
  // D = A(16x32 f16) * B(32x16 f16) + C(16x16 f32)
  return __builtin_amdgcn_wmma_f32_16x16x32_f16(
      /*neg_a=*/false, a, /*neg_b=*/false, b,
      /*c_mod=*/(short)0, c, /*reuse_a=*/false, /*reuse_b=*/false);
}

__global__ __launch_bounds__(256) void VolumeSDFRenderer_kernel(
    const float* __restrict__ dist,    // [R,128]
    const float* __restrict__ color,   // [R,128,3]
    const float* __restrict__ depth,   // [R,128]
    float* __restrict__ out,           // [R,3]
    int n_rays)
{
  __shared__ float sdbuf_all[WAVES_PER_BLOCK * RAYS_PER_WAVE * LDS_P];

  const int lane = threadIdx.x & 31;
  const int wave = threadIdx.x >> 5;
  float* sb = sdbuf_all + wave * (RAYS_PER_WAVE * LDS_P);

  const int ray0 = (blockIdx.x * WAVES_PER_BLOCK + wave) * RAYS_PER_WAVE;
  if (ray0 + RAYS_PER_WAVE > n_rays) return;   // whole-wave exit (EXEC stays all-1 for WMMA)

  const int hi  = lane >> 4;   // 0: lanes 0-15, 1: lanes 16-31
  const int l15 = lane & 15;

  // ------------------------------------------------------------------
  // Constant B fragments (32x16 f16, wave32 layout):
  //   VGPR g holds K = 2g+8*(g>=4),+1 (lanes 0-15) / 8+2g+8*(g>=4),+1 (lanes 16-31),
  //   column N = lane & 15.
  //   b_diag[k,n] = (k <  n)      -> same-chunk strict prefix
  //   b_mid [k,n] = (k < 16 + n)  -> chunk shifted 16 below the columns
  //   b_ones[k,n] = 1             -> full-chunk row sums (running prefix)
  // ------------------------------------------------------------------
  v16h b_diag, b_mid, b_ones;
  #pragma unroll
  for (int g = 0; g < 8; ++g) {
    #pragma unroll
    for (int h = 0; h < 2; ++h) {
      int Kl = ((g & 4) << 2) + (hi ? 8 : 0) + 2 * (g & 3) + h;
      b_diag[2*g + h] = (_Float16)((Kl < l15)      ? 1.0f : 0.0f);
      b_mid [2*g + h] = (_Float16)((Kl < 16 + l15) ? 1.0f : 0.0f);
      b_ones[2*g + h] = (_Float16)1.0f;
    }
  }

  // ------------------------------------------------------------------
  // Phase 1: coalesced compute of sd = density * delta, staged to LDS.
  // flat = m*128 + n over the wave's 16 rays; 64 steps of 32 lanes.
  // ------------------------------------------------------------------
  const long base = (long)ray0 * NSAMP;
  for (int step = 0; step < 64; ++step) {
    int flat = step * 32 + lane;
    int m = flat >> 7;
    int n = flat & 127;
    float d   = dist[base + flat];
    float e   = __expf(-SDF_S * d);
    float inv = __builtin_amdgcn_rcpf(1.0f + e);
    float density = SDF_S * e * inv * inv;
    float dl  = depth[base + flat];
    float delta;
    if (n == 127) {
      delta = 1e10f;                              // terminal FAR_DELTA
    } else {
      delta = depth[base + flat + 1] - dl;        // same/adjacent cacheline
    }
    sb[m * LDS_P + n] = density * delta;
  }
  // Per-wave LDS region + in-order DS pipeline: no block barrier needed.

  // ------------------------------------------------------------------
  // Phase 2: tau via WMMA prefix-matmul, immediately consumed into the
  // per-ray weighted color accumulators (C-tile layout: VGPR g holds
  // row m = g + 8*hi, column n = l15 + 16*ic).
  // ------------------------------------------------------------------
  float acc[24];
  #pragma unroll
  for (int i = 0; i < 24; ++i) acc[i] = 0.0f;

  auto consume = [&](const v8f& t, int nbase) {
    int n = nbase + l15;
    #pragma unroll
    for (int g = 0; g < 8; ++g) {
      int m = g + hi * 8;
      float s = sb[m * LDS_P + n];                      // sd in f32
      float w = __expf(-t[g]) * (1.0f - __expf(-s));    // exp(-tau)*(1-exp(-sd))
      const float* cp = color + ((long)(ray0 + m) * NSAMP + n) * 3;
      acc[g*3 + 0] = fmaf(w, cp[0], acc[g*3 + 0]);
      acc[g*3 + 1] = fmaf(w, cp[1], acc[g*3 + 1]);
      acc[g*3 + 2] = fmaf(w, cp[2], acc[g*3 + 2]);
    }
  };

  v8f R = {};                                  // running full-chunk prefix
  const float* rowp = sb + l15 * LDS_P;        // this lane's A-matrix row
  #pragma unroll
  for (int j = 0; j < 4; ++j) {                // K chunks of 32
    // A fragment (16x32 f16): VGPR g -> K = kb + (g>=4?16:0) + 2*(g&3) (+1)
    v16h a;
    int kb = 32 * j + (hi ? 8 : 0);
    #pragma unroll
    for (int g = 0; g < 8; ++g) {
      int K = kb + ((g & 4) << 2) + 2 * (g & 3);
      // clamp: terminal sd can be ~5e10 -> f16 Inf -> Inf*0=NaN in WMMA.
      a[2*g]     = (_Float16)fminf(rowp[K],     60000.0f);
      a[2*g + 1] = (_Float16)fminf(rowp[K + 1], 60000.0f);
    }
    v8f t_even = wmma16x16x32(a, b_diag, R);   // columns 32j     .. 32j+15
    consume(t_even, 32 * j);
    v8f t_odd  = wmma16x16x32(a, b_mid,  R);   // columns 32j+16  .. 32j+31
    consume(t_odd, 32 * j + 16);
    if (j < 3) R = wmma16x16x32(a, b_ones, R); // accumulate full-chunk sums
  }

  // ------------------------------------------------------------------
  // Reduce over the 16 columns held within each 16-lane half.
  // ------------------------------------------------------------------
  #pragma unroll
  for (int i = 0; i < 24; ++i) {
    float v = acc[i];
    v += __shfl_xor(v, 1);
    v += __shfl_xor(v, 2);
    v += __shfl_xor(v, 4);
    v += __shfl_xor(v, 8);
    acc[i] = v;
  }
  if (l15 == 0) {                 // lane 0 -> rays m=0..7, lane 16 -> m=8..15
    int rbase = ray0 + hi * 8;
    #pragma unroll
    for (int g = 0; g < 8; ++g) {
      #pragma unroll
      for (int c = 0; c < 3; ++c) {
        float v = acc[g*3 + c];
        out[(long)(rbase + g) * 3 + c] = fminf(fmaxf(v, 0.0f), 1.0f);
      }
    }
  }
}

extern "C" void kernel_launch(void* const* d_in, const int* in_sizes, int n_in,
                              void* d_out, int out_size, void* d_ws, size_t ws_size,
                              hipStream_t stream) {
  (void)n_in; (void)out_size; (void)d_ws; (void)ws_size;
  const float* dist  = (const float*)d_in[0];   // [R,128]
  const float* color = (const float*)d_in[1];   // [R,128,3]
  const float* depth = (const float*)d_in[2];   // [R,128]
  float* out = (float*)d_out;                   // [R,3]
  int n_rays = in_sizes[0] / NSAMP;             // 262144
  int blocks = (n_rays + RAYS_PER_BLOCK - 1) / RAYS_PER_BLOCK;  // 2048
  VolumeSDFRenderer_kernel<<<blocks, 256, 0, stream>>>(dist, color, depth, out, n_rays);
}
